// SelfAttention_20478404067999
// MI455X (gfx1250) — matrix operand order repaired and checked by reading, not compiled, not run on hardware
//
#include <hip/hip_runtime.h>
#include <hip/hip_bf16.h>

typedef __attribute__((ext_vector_type(16))) _Float16 v16h;
typedef __attribute__((ext_vector_type(8)))  _Float16 v8h;
typedef __attribute__((ext_vector_type(8)))  float    v8f;
typedef int v4i_vs __attribute__((vector_size(16)));   // matches builtin param

#define EMBED  1024
#define NHEADS 16
#define HDIM   64
#define BATCH  4
#define SEQ    2048
#define ROWS   (BATCH * SEQ)   // 8192

// ---------------------------------------------------------------------------
// gfx1250 async global->LDS copy (ASYNCcnt path), guarded so the file still
// compiles on toolchains lacking the builtins. Signature probe-confirmed:
// (int4 addrspace(1)*, int4 addrspace(3)*, imm offset, imm cpol).
// ---------------------------------------------------------------------------
#if __has_builtin(__builtin_amdgcn_global_load_async_to_lds_b128) && \
    __has_builtin(__builtin_amdgcn_s_wait_asynccnt)
#define HAVE_ASYNC_LDS 1
__device__ __forceinline__ void copy16_g2l(const _Float16* g, _Float16* l) {
  // generic LDS pointer low 32 bits == LDS offset (flat aperture rules)
  __builtin_amdgcn_global_load_async_to_lds_b128(
      (__attribute__((address_space(1))) v4i_vs*)(unsigned long long)g,
      (__attribute__((address_space(3))) v4i_vs*)(unsigned)(unsigned long long)l,
      0, 0);
}
__device__ __forceinline__ void async_fence() {
  __builtin_amdgcn_s_wait_asynccnt(0);
}
#else
#define HAVE_ASYNC_LDS 0
__device__ __forceinline__ void copy16_g2l(const _Float16* g, _Float16* l) {
  *(v8h*)l = *(const v8h*)g;
}
__device__ __forceinline__ void async_fence() {}
#endif

// ---------------------------------------------------------------------------
// WMMA helpers (CDNA5 16x16x32 f16 -> f32)
// ---------------------------------------------------------------------------
__device__ __forceinline__ v16h frag_join(v8h lo, v8h hi) {
  return __builtin_shufflevector(lo, hi, 0, 1, 2, 3, 4, 5, 6, 7,
                                         8, 9, 10, 11, 12, 13, 14, 15);
}

// A-fragment 16x32 f16 (row = lane&15): lane half hv holds
// K = hv*8..hv*8+7 (VGPR0-3) and K = 16+hv*8..23+hv*8 (VGPR4-7).
__device__ __forceinline__ v16h load_frag_a(const _Float16* p, int ld) {
  const int lane = threadIdx.x & 31;
  const int m  = lane & 15;
  const int hv = lane >> 4;
  const _Float16* r = p + m * ld + hv * 8;
  v8h lo = *(const v8h*)(r);
  v8h hi = *(const v8h*)(r + 16);
  return frag_join(lo, hi);
}

// B-fragment 32x16 f16 (col = lane&15): lane half hv holds
// K = hv*8..hv*8+7 and K = 16+hv*8..23+hv*8 of its column (column stored contiguously).
__device__ __forceinline__ v16h load_frag_b(const _Float16* p, int ld) {
  const int lane = threadIdx.x & 31;
  const int n  = lane & 15;
  const int hv = lane >> 4;
  const _Float16* r = p + n * ld + hv * 8;
  v8h lo = *(const v8h*)(r);
  v8h hi = *(const v8h*)(r + 16);
  return frag_join(lo, hi);
}

__device__ __forceinline__ v8f wmma_16x16x32(v16h a, v16h b, v8f c) {
  return __builtin_amdgcn_wmma_f32_16x16x32_f16(false, a, false, b,
                                                (short)0, c, false, false);
}

// ---------------------------------------------------------------------------
// fp32 -> f16 convert
// ---------------------------------------------------------------------------
__global__ void cvt_h_kernel(const float* __restrict__ in,
                             _Float16* __restrict__ out, int n) {
  int i = (blockIdx.x * blockDim.x + threadIdx.x) * 4;
  if (i + 3 < n) {
    float4 f = *(const float4*)(in + i);
    out[i + 0] = (_Float16)f.x;
    out[i + 1] = (_Float16)f.y;
    out[i + 2] = (_Float16)f.z;
    out[i + 3] = (_Float16)f.w;
  }
}

// fp32 [K=1024][N=1024] -> f16 transposed [N][K]
__global__ void cvt_t_kernel(const float* __restrict__ W,
                             _Float16* __restrict__ Wt) {
  int idx = blockIdx.x * blockDim.x + threadIdx.x;   // 0..1M-1
  int k = idx >> 10;
  int n = idx & 1023;
  Wt[n * EMBED + k] = (_Float16)W[idx];
}

// ---------------------------------------------------------------------------
// WMMA GEMM: C[M,N] = (A[M,K] @ Bt[N,K]^T + bias) * scale
// 256 threads (8 waves), 128x128 tile, BK=64; wave computes 32x64
// (16 WMMAs per barrier pair, fragments preloaded before the MMA chain).
// ---------------------------------------------------------------------------
#define BM  128
#define BN  128
#define BK  64
#define LDT 72   // padded LDS leading dim (16B-aligned rows)

__global__ __launch_bounds__(256)
void gemm_wmma_kernel(const _Float16* __restrict__ A,
                      const _Float16* __restrict__ Bt,
                      const float* __restrict__ bias,
                      _Float16* __restrict__ outH,
                      float* __restrict__ outF,
                      int M, int N, int K, float scale) {
  __shared__ __attribute__((aligned(16))) _Float16 As[BM * LDT];  // 18 KiB
  __shared__ __attribute__((aligned(16))) _Float16 Bs[BN * LDT];  // 18 KiB

  const int tid  = threadIdx.x;
  const int lane = tid & 31;
  const int wave = tid >> 5;      // 0..7
  const int wr   = wave >> 1;     // 0..3  (row group of 32)
  const int wc   = wave & 1;      // 0..1  (col group of 64)
  const int nL   = lane & 15;
  const int hv   = lane >> 4;

  const int nTilesN = N / BN;
  const int m0 = (blockIdx.x / nTilesN) * BM;
  const int n0 = (blockIdx.x % nTilesN) * BN;

  v8f acc[2][4];
#pragma unroll
  for (int mi = 0; mi < 2; ++mi)
#pragma unroll
    for (int ni = 0; ni < 4; ++ni)
      acc[mi][ni] = (v8f){0.f, 0.f, 0.f, 0.f, 0.f, 0.f, 0.f, 0.f};

  for (int k0 = 0; k0 < K; k0 += BK) {
    // Stage 128x64 A tile and 128x64 Bt tile: 1024 chunks of 8 halves each.
#pragma unroll
    for (int i = 0; i < 4; ++i) {
      int c   = tid + 256 * i;     // 0..1023
      int row = c >> 3;            // 0..127
      int kc  = (c & 7) * 8;       // 0..56
      copy16_g2l(&A[(size_t)(m0 + row) * K + k0 + kc],  &As[row * LDT + kc]);
      copy16_g2l(&Bt[(size_t)(n0 + row) * K + k0 + kc], &Bs[row * LDT + kc]);
    }
    if (k0 + BK < K) {   // hint next k-slice into cache (global_prefetch_b8)
      __builtin_prefetch(&A[(size_t)(m0 + (tid >> 1)) * K + k0 + BK], 0, 1);
      __builtin_prefetch(&Bt[(size_t)(n0 + (tid >> 1)) * K + k0 + BK], 0, 1);
    }
    async_fence();
    __syncthreads();

#pragma unroll
    for (int kk = 0; kk < BK; kk += 32) {
      // Preload all fragments, then issue the 8-WMMA chain back-to-back.
      v16h a0 = load_frag_a(&As[(32 * wr +  0) * LDT + kk], LDT);
      v16h a1 = load_frag_a(&As[(32 * wr + 16) * LDT + kk], LDT);
      v16h bfr[4];
#pragma unroll
      for (int ni = 0; ni < 4; ++ni)
        bfr[ni] = load_frag_b(&Bs[(64 * wc + 16 * ni) * LDT + kk], LDT);
#pragma unroll
      for (int ni = 0; ni < 4; ++ni) {
        acc[0][ni] = wmma_16x16x32(a0, bfr[ni], acc[0][ni]);
        acc[1][ni] = wmma_16x16x32(a1, bfr[ni], acc[1][ni]);
      }
    }
    __syncthreads();
  }

#pragma unroll
  for (int mi = 0; mi < 2; ++mi)
#pragma unroll
    for (int ni = 0; ni < 4; ++ni)
#pragma unroll
      for (int v = 0; v < 8; ++v) {
        int row = m0 + 32 * wr + 16 * mi + v + 8 * hv;
        int col = n0 + 64 * wc + 16 * ni + nL;
        float val = (acc[mi][ni][v] + bias[col]) * scale;
        if (outH) outH[(size_t)row * N + col] = (_Float16)val;
        else      outF[(size_t)row * N + col] = val;
      }
}

// ---------------------------------------------------------------------------
// Flash attention: block = (b, h, 64 query rows); 128 threads = 4 waves,
// each wave owns 16 query rows. Online softmax, causal mask.
// Q is pre-scaled by 1/sqrt(Dh) in its projection GEMM.
// ---------------------------------------------------------------------------
#define LDA 72   // padded LDS leading dim

__global__ __launch_bounds__(128)
void attn_kernel(const _Float16* __restrict__ Q,
                 const _Float16* __restrict__ Kh,
                 const _Float16* __restrict__ Vh,
                 _Float16* __restrict__ Y) {
  __shared__ __attribute__((aligned(16))) _Float16 Ks[64 * LDA];
  __shared__ __attribute__((aligned(16))) _Float16 Vt[64 * LDA];
  __shared__ __attribute__((aligned(16))) _Float16 Ps[4][16 * LDA];

  const int bh = blockIdx.x;            // 0..63
  const int b  = bh / NHEADS;
  const int h  = bh % NHEADS;
  const int qb = blockIdx.y;            // 0..31  (64-row query block)
  const int tid  = threadIdx.x;
  const int lane = tid & 31;
  const int wave = tid >> 5;            // 0..3
  const int nL = lane & 15;
  const int hv = lane >> 4;

  const size_t rowBase = (size_t)b * SEQ;
  const int    colBase = h * HDIM;

  // Per-wave Q fragments (rows qb*64 + wave*16 .. +15, d = 0..63)
  v16h qf[2];
  {
    const _Float16* qp = Q + (rowBase + qb * 64 + wave * 16) * EMBED + colBase;
    qf[0] = load_frag_a(qp, EMBED);
    qf[1] = load_frag_a(qp + 32, EMBED);
  }

  v8f o[4];
#pragma unroll
  for (int dt = 0; dt < 4; ++dt)
    o[dt] = (v8f){0.f, 0.f, 0.f, 0.f, 0.f, 0.f, 0.f, 0.f};
  float mrow[8], lrow[8];
#pragma unroll
  for (int v = 0; v < 8; ++v) { mrow[v] = -1e30f; lrow[v] = 0.f; }

  for (int j = 0; j <= qb; ++j) {
    const _Float16* kp = Kh + (rowBase + j * 64) * EMBED + colBase;
    const _Float16* vp = Vh + (rowBase + j * 64) * EMBED + colBase;
#pragma unroll
    for (int i = 0; i < 4; ++i) {
      int c  = tid + 128 * i;     // 0..511
      int r  = c >> 3;            // 0..63
      int kc = (c & 7) * 8;       // 0..56
      copy16_g2l(kp + (size_t)r * EMBED + kc, &Ks[r * LDA + kc]);
      v8h vv = *(const v8h*)(vp + (size_t)r * EMBED + kc);
#pragma unroll
      for (int e = 0; e < 8; ++e) Vt[(kc + e) * LDA + r] = vv[e];  // transpose
    }
    if (j < qb) {  // prefetch next K/V block
      __builtin_prefetch(kp + (size_t)(64 + (tid >> 1)) * EMBED, 0, 1);
      __builtin_prefetch(vp + (size_t)(64 + (tid >> 1)) * EMBED, 0, 1);
    }
    async_fence();
    __syncthreads();

    // S = Q @ K^T : 16 x 64 per wave (4 accumulators), K-dim = d = 64
    v8f s[4];
#pragma unroll
    for (int t = 0; t < 4; ++t) {
      s[t] = (v8f){0.f, 0.f, 0.f, 0.f, 0.f, 0.f, 0.f, 0.f};
      v16h b0 = load_frag_b(&Ks[(t * 16) * LDA], LDA);       // d = 0..31
      v16h b1 = load_frag_b(&Ks[(t * 16) * LDA + 32], LDA);  // d = 32..63
      s[t] = wmma_16x16x32(qf[0], b0, s[t]);
      s[t] = wmma_16x16x32(qf[1], b1, s[t]);
    }

    // Causal mask on the diagonal block
    if (j == qb) {
#pragma unroll
      for (int t = 0; t < 4; ++t)
#pragma unroll
        for (int v = 0; v < 8; ++v) {
          int qrow = wave * 16 + v + 8 * hv;
          int scol = t * 16 + nL;
          if (scol > qrow) s[t][v] = -1e30f;
        }
    }

    // Online softmax per row (row = (v, lane-half); reduce across 16 lanes)
#pragma unroll
    for (int v = 0; v < 8; ++v) {
      float bm = fmaxf(fmaxf(s[0][v], s[1][v]), fmaxf(s[2][v], s[3][v]));
      bm = fmaxf(bm, __shfl_xor(bm, 1, 32));
      bm = fmaxf(bm, __shfl_xor(bm, 2, 32));
      bm = fmaxf(bm, __shfl_xor(bm, 4, 32));
      bm = fmaxf(bm, __shfl_xor(bm, 8, 32));
      float mnew  = fmaxf(mrow[v], bm);
      float alpha = __expf(mrow[v] - mnew);
      mrow[v] = mnew;
      float rs = 0.f;
#pragma unroll
      for (int t = 0; t < 4; ++t) {
        float p = __expf(s[t][v] - mnew);
        s[t][v] = p;
        rs += p;
      }
      rs += __shfl_xor(rs, 1, 32);
      rs += __shfl_xor(rs, 2, 32);
      rs += __shfl_xor(rs, 4, 32);
      rs += __shfl_xor(rs, 8, 32);
      lrow[v] = lrow[v] * alpha + rs;
#pragma unroll
      for (int dt = 0; dt < 4; ++dt) o[dt][v] *= alpha;
    }

    // Bounce P through per-wave LDS to reshape C/D layout -> A-fragment layout
    _Float16* pw = &Ps[wave][0];
#pragma unroll
    for (int t = 0; t < 4; ++t)
#pragma unroll
      for (int v = 0; v < 8; ++v)
        pw[(v + 8 * hv) * LDA + t * 16 + nL] = (_Float16)s[t][v];

    // O += P @ V  (K-dim = s = 64; B-frags from transposed V tile)
    v16h p0 = load_frag_a(pw, LDA);
    v16h p1 = load_frag_a(pw + 32, LDA);
#pragma unroll
    for (int dt = 0; dt < 4; ++dt) {
      v16h b0 = load_frag_b(&Vt[(dt * 16) * LDA], LDA);
      v16h b1 = load_frag_b(&Vt[(dt * 16) * LDA + 32], LDA);
      o[dt] = wmma_16x16x32(p0, b0, o[dt]);
      o[dt] = wmma_16x16x32(p1, b1, o[dt]);
    }
    __syncthreads();
  }

  // Normalize and store Y (f16, [ROWS, EMBED])
  const size_t orow0 = rowBase + qb * 64 + wave * 16;
#pragma unroll
  for (int dt = 0; dt < 4; ++dt)
#pragma unroll
    for (int v = 0; v < 8; ++v) {
      int r = v + 8 * hv;
      float val = o[dt][v] / lrow[v];
      Y[(orow0 + r) * EMBED + colBase + dt * 16 + nL] = (_Float16)val;
    }
}

// ---------------------------------------------------------------------------
extern "C" void kernel_launch(void* const* d_in, const int* in_sizes, int n_in,
                              void* d_out, int out_size, void* d_ws,
                              size_t ws_size, hipStream_t stream) {
  const float* x  = (const float*)d_in[0];
  const float* Wq = (const float*)d_in[1];
  const float* bq = (const float*)d_in[2];
  const float* Wk = (const float*)d_in[3];
  const float* bk = (const float*)d_in[4];
  const float* Wv = (const float*)d_in[5];
  const float* bv = (const float*)d_in[6];
  const float* Wo = (const float*)d_in[7];
  const float* bo = (const float*)d_in[8];
  float* out = (float*)d_out;

  // Workspace layout (f16 buffers), ~88 MiB total
  char* p = (char*)d_ws;
  const size_t XH  = (size_t)ROWS * EMBED * sizeof(_Float16);   // 16 MiB
  const size_t WT  = (size_t)EMBED * EMBED * sizeof(_Float16);  //  2 MiB
  _Float16* xh  = (_Float16*)p;              p += XH;
  _Float16* Wqt = (_Float16*)p;              p += WT;
  _Float16* Wkt = (_Float16*)p;              p += WT;
  _Float16* Wvt = (_Float16*)p;              p += WT;
  _Float16* Wot = (_Float16*)p;              p += WT;
  _Float16* Qh  = (_Float16*)p;              p += XH;
  _Float16* Kh  = (_Float16*)p;              p += XH;
  _Float16* Vh  = (_Float16*)p;              p += XH;
  _Float16* Yh  = (_Float16*)p;              p += XH;

  const int nx = ROWS * EMBED;                               // 8M elements
  cvt_h_kernel<<<nx / (256 * 4), 256, 0, stream>>>(x, xh, nx);

  const int nw = EMBED * EMBED;                              // 1M elements
  cvt_t_kernel<<<nw / 256, 256, 0, stream>>>(Wq, Wqt);
  cvt_t_kernel<<<nw / 256, 256, 0, stream>>>(Wk, Wkt);
  cvt_t_kernel<<<nw / 256, 256, 0, stream>>>(Wv, Wvt);
  cvt_t_kernel<<<nw / 256, 256, 0, stream>>>(Wo, Wot);

  const int gemmGrid = (ROWS / BM) * (EMBED / BN);           // 512 blocks
  const float qscale = 0.125f;                               // 1/sqrt(64)
  gemm_wmma_kernel<<<gemmGrid, 256, 0, stream>>>(xh, Wqt, bq, Qh, nullptr,
                                                 ROWS, EMBED, EMBED, qscale);
  gemm_wmma_kernel<<<gemmGrid, 256, 0, stream>>>(xh, Wkt, bk, Kh, nullptr,
                                                 ROWS, EMBED, EMBED, 1.0f);
  gemm_wmma_kernel<<<gemmGrid, 256, 0, stream>>>(xh, Wvt, bv, Vh, nullptr,
                                                 ROWS, EMBED, EMBED, 1.0f);

  dim3 agrid(BATCH * NHEADS, SEQ / 64);                      // (64, 32)
  attn_kernel<<<agrid, 128, 0, stream>>>(Qh, Kh, Vh, Yh);

  gemm_wmma_kernel<<<gemmGrid, 256, 0, stream>>>(Yh, Wot, bo, nullptr, out,
                                                 ROWS, EMBED, EMBED, 1.0f);
}
